// TurboQuantMSE_61838939127829
// MI455X (gfx1250) — matrix operand order repaired and checked by reading, not compiled, not run on hardware
//
#include <hip/hip_runtime.h>
#include <hip/hip_bf16.h>

typedef __attribute__((ext_vector_type(16))) _Float16 v16h;
typedef __attribute__((ext_vector_type(8)))  _Float16 v8h;
typedef __attribute__((ext_vector_type(8)))  float    v8f;
typedef __attribute__((ext_vector_type(4)))  unsigned int u32x4;
typedef __attribute__((ext_vector_type(8)))  int      i32x8;
typedef __attribute__((ext_vector_type(4)))  int      i32x4;

#define DIM 128
// 128 fragments total (64 fwd + 64 bwd), each laid out as
// [chunk c=0/1][lane=32][8 halfs] = 1024 B  ->  128 KB total.
#define NFRAG      128
#define FRAG_HALFS 512
#define WS_HALFS   (NFRAG * FRAG_HALFS)      // 65536 halfs = 128 KB

// ---------------------------------------------------------------------------
// Prep: build M_f = (1/16) H D1 H D0 (cols 0..127) and M_b = (1/4096) D0 H D1 H
// (rows 0..127) in f16, pre-swizzled into WMMA A-layout fragments stored as
// [frag][chunk][lane][8] (16B per lane per chunk -> conflict-free ds_load_b128).
//   A(m,k): lane = m + 16*(k>=16), half h: k = h + 16*(lane>=16)
// Uses (H D1 H)(i,j) = F(i^j) with F = unnormalized FWHT of s1.
// ---------------------------------------------------------------------------
__global__ __launch_bounds__(256) void tq_prep(const float* __restrict__ signs,
                                               _Float16* __restrict__ wsH) {
  __shared__ float F[256];
  const int tid = threadIdx.x;
  const float* s1 = signs + 256;
  float acc = 0.f;
  for (int t = 0; t < 256; ++t) {
    float v = s1[t];
    acc += (__builtin_popcount(tid & t) & 1) ? -v : v;
  }
  F[tid] = acc;
  __syncthreads();

  const int e    = blockIdx.x * 256 + tid;     // 65536 entries total
  const int q    = e & 7;                      // half-within-chunk
  const int lane = (e >> 3) & 31;
  const int c    = (e >> 8) & 1;               // 16B chunk (halves 0-7 / 8-15)
  const int f    = e >> 9;                     // fragment id 0..127
  const int m = lane & 15, g = lane >> 4;
  const int k = q + 8 * c + 16 * g;            // K index within 32-wide slice

  float val;
  if (f < 64) {                                // fwd: f = nn*4 + kk
    const int nn = f >> 2, kk = f & 3;
    const int i = 16 * nn + m;                 // row of M_f (y-index)
    const int j = 32 * kk + k;                 // col of M_f (input col 0..127)
    val = (1.f / 16.f) * signs[j] * F[i ^ j];
  } else {                                     // bwd: f-64 = oo*8 + kk
    const int f2 = f - 64;
    const int oo = f2 >> 3, kk = f2 & 7;
    const int i = 16 * oo + m;                 // row of M_b (out col 0..127)
    const int j = 32 * kk + k;                 // col of M_b (y-index 0..255)
    val = (1.f / 4096.f) * signs[i] * F[i ^ j];
  }
  wsH[e] = (_Float16)val;
}

// Branchless 16-level quantize->dequantize (levels are symmetric).
__device__ __forceinline__ float dq(float y) {
  float a = fabsf(y);
  float pv = 0.1284f;
  pv = a > 0.25825f  ? 0.3881f : pv;
  pv = a > 0.52245f  ? 0.6568f : pv;
  pv = a > 0.7996f   ? 0.9424f : pv;
  pv = a > 1.0993f   ? 1.2562f : pv;
  pv = a > 1.43715f  ? 1.6181f : pv;
  pv = a > 1.84355f  ? 2.069f  : pv;
  pv = a > 2.4008f   ? 2.7326f : pv;
  return copysignf(pv, y);
}

__device__ __forceinline__ v16h ldsFrag(const _Float16* shA, int f, int lane) {
  const v8h* p = (const v8h*)(shA + f * FRAG_HALFS + lane * 8);
  v8h lo = p[0];                               // halves 0..7  (+lane*16 B)
  v8h hi = p[32];                              // halves 8..15 (+256 halfs)
  return __builtin_shufflevector(lo, hi, 0, 1, 2, 3, 4, 5, 6, 7,
                                         8, 9, 10, 11, 12, 13, 14, 15);
}

// ---------------------------------------------------------------------------
// Main: TDM-stage the 128 KB of A-fragments to LDS once per block, then each
// wave processes a 16-row tile:
//   Forward : Y^T(256x16) = M_f(256x128) * Xu^T(128x16)   -> 64 WMMA
//   Quantize: elementwise on C/D fragments
//   Backward: Out^T(128x16) = M_b(128x256) * Q^T(256x16)  -> 64 WMMA
// ---------------------------------------------------------------------------
__global__ __launch_bounds__(256) void tq_main(const float* __restrict__ x,
                                               const _Float16* __restrict__ wsH,
                                               float* __restrict__ out,
                                               int ntiles) {
  extern __shared__ _Float16 shA[];            // 128 KB dynamic LDS

#if __has_builtin(__builtin_amdgcn_tensor_load_to_lds)
  if (threadIdx.x < 32) {                      // wave 0 drives the TDM
    const unsigned long long ga = (unsigned long long)(uintptr_t)wsH;
    const unsigned int lds_off = (unsigned int)(uintptr_t)&shA[0]; // low 32 = LDS offset
    u32x4 g0;
    g0[0] = 1u;                                            // count=1, user mode
    g0[1] = lds_off;                                       // lds_addr
    g0[2] = (unsigned int)(ga & 0xFFFFFFFFu);              // global_addr[31:0]
    g0[3] = (unsigned int)((ga >> 32) & 0x01FFFFFFu) | (2u << 30); // addr[56:32], type=2
    i32x8 g1;
    g1[0] = 3 << 16;                                       // mask=0, data_size=8B
    g1[1] = 0x4000 << 16;                                  // tensor_dim0 = 16384 (lo16)
    g1[2] = 1 << 16;                                       // dim0 hi=0, tensor_dim1 = 1
    g1[3] = 0x4000 << 16;                                  // dim1 hi=0, tile_dim0 = 16384
    g1[4] = 1;                                             // tile_dim1 = 1, tile_dim2 = 0
    g1[5] = 16384;                                         // tensor_dim0_stride lo
    g1[6] = 0;                                             // stride0 hi, stride1 lo
    g1[7] = 0;
    i32x4 gz4 = {0, 0, 0, 0};
    i32x8 gz8 = {0, 0, 0, 0, 0, 0, 0, 0};
    __builtin_amdgcn_tensor_load_to_lds(g0, g1, gz4, gz4, gz8, 0);
    __builtin_amdgcn_s_wait_tensorcnt(0);
  }
#else
  {                                            // fallback: cooperative copy
    const float4* src = (const float4*)wsH;
    float4* dst = (float4*)shA;
    #pragma unroll
    for (int i = 0; i < 32; ++i)
      dst[threadIdx.x + 256 * i] = src[threadIdx.x + 256 * i];
  }
#endif
  __syncthreads();

  const int lane = threadIdx.x & 31;
  const int wave = threadIdx.x >> 5;
  const int tile = blockIdx.x * 8 + wave;
  if (tile >= ntiles) return;                  // wave-uniform: EXEC stays full
  const int m = lane & 15, g = lane >> 4;
  const size_t row = (size_t)tile * 16 + m;
  const float* xr = x + row * DIM;

  // Load row halves straight into B layout order; accumulate sum of squares.
  // B(k_local, n): lane = n + 16*(k_local>=16), half = k_local%16.
  float xv[4][16];
  float ss = 0.f;
  #pragma unroll
  for (int kk = 0; kk < 4; ++kk) {
    const float4* p4 = (const float4*)(xr + 32 * kk + 16 * g);
    #pragma unroll
    for (int q4 = 0; q4 < 4; ++q4) {
      float4 v = p4[q4];
      xv[kk][4 * q4 + 0] = v.x; xv[kk][4 * q4 + 1] = v.y;
      xv[kk][4 * q4 + 2] = v.z; xv[kk][4 * q4 + 3] = v.w;
      ss += v.x * v.x + v.y * v.y + v.z * v.z + v.w * v.w;
    }
  }
  ss += __shfl_xor(ss, 16, 32);                // lanes m and m+16 hold row m
  const float norm = sqrtf(ss);
  const float rn = 1.f / (norm + 1e-10f);

  v16h Bx[4];
  #pragma unroll
  for (int kk = 0; kk < 4; ++kk)
    #pragma unroll
    for (int h = 0; h < 16; ++h)
      Bx[kk][h] = (_Float16)(xv[kk][h] * rn);

  // Forward GEMM + quantize. Qf[nn] holds deq(Y^T) chunk nn in C/D layout.
  v8f Qf[16];
  #pragma unroll
  for (int nn = 0; nn < 16; ++nn) {
    v8f c = {0.f, 0.f, 0.f, 0.f, 0.f, 0.f, 0.f, 0.f};
    #pragma unroll
    for (int kk = 0; kk < 4; ++kk) {
      v16h a = ldsFrag(shA, nn * 4 + kk, lane);
      c = __builtin_amdgcn_wmma_f32_16x16x32_f16(false, a, false, Bx[kk],
                                                 (short)0, c, false, false);
    }
    v8f qv;
    #pragma unroll
    for (int r = 0; r < 8; ++r) qv[r] = dq(c[r]);
    Qf[nn] = qv;
  }

  // C/D -> B layout: C/D splits rows at M=8, B splits K at 16 -> xor16 + select.
  v16h Bq[8];
  #pragma unroll
  for (int kk = 0; kk < 8; ++kk) {
    v16h b;
    #pragma unroll
    for (int r = 0; r < 8; ++r) {
      float a0 = Qf[2 * kk][r];
      float a1 = Qf[2 * kk + 1][r];
      float x0 = __shfl_xor(a0, 16, 32);
      float x1 = __shfl_xor(a1, 16, 32);
      b[r]     = (_Float16)(g ? x1 : a0);      // halves 0..7
      b[8 + r] = (_Float16)(g ? a1 : x0);      // halves 8..15
    }
    Bq[kk] = b;
  }

  // Backward GEMM, scale by norm, store Out^T fragments (8 contiguous f32/lane).
  float* orow = out + row * DIM;
  #pragma unroll
  for (int oo = 0; oo < 8; ++oo) {
    v8f c = {0.f, 0.f, 0.f, 0.f, 0.f, 0.f, 0.f, 0.f};
    #pragma unroll
    for (int kk = 0; kk < 8; ++kk) {
      v16h a = ldsFrag(shA, 64 + oo * 8 + kk, lane);
      c = __builtin_amdgcn_wmma_f32_16x16x32_f16(false, a, false, Bq[kk],
                                                 (short)0, c, false, false);
    }
    float4 o0, o1;
    o0.x = c[0] * norm; o0.y = c[1] * norm; o0.z = c[2] * norm; o0.w = c[3] * norm;
    o1.x = c[4] * norm; o1.y = c[5] * norm; o1.z = c[6] * norm; o1.w = c[7] * norm;
    float* dst = orow + 16 * oo + 8 * g;       // out col = 16*oo + 8*g + r
    ((float4*)dst)[0] = o0;
    ((float4*)dst)[1] = o1;
  }
}

extern "C" void kernel_launch(void* const* d_in, const int* in_sizes, int n_in,
                              void* d_out, int out_size, void* d_ws, size_t ws_size,
                              hipStream_t stream) {
  const float* x     = (const float*)d_in[0];
  const float* signs = (const float*)d_in[1];
  _Float16*    wsH   = (_Float16*)d_ws;        // needs 128 KB
  float*       outp  = (float*)d_out;

  const int rows   = in_sizes[0] / DIM;        // 524288
  const int ntiles = rows / 16;                // 32768
  const int blocks = (ntiles + 7) / 8;         // 4096 (8 waves/block)

  tq_prep<<<256, 256, 0, stream>>>(signs, wsH);
  tq_main<<<blocks, 256, (size_t)(WS_HALFS * sizeof(_Float16)), stream>>>(
      x, wsH, outp, ntiles);
}
